// KPConv_42090679501115
// MI455X (gfx1250) — matrix operand order repaired and checked by reading, not compiled, not run on hardware
//
#include <hip/hip_runtime.h>

// KPConv fused kernel for gfx1250 (MI455X), wave32 + v_wmma_f32_16x16x32_f16.
// Requires ws_size >= 131072 bytes (f16 pre-swizzled weight tensor).

#define HN   32      // neighbors per query
#define KP   15      // kernel points
#define CIN  64
#define COUT 64
#define QT   16      // queries per workgroup tile
#define INV_SIGMA 10.0f
#define PT_INF 1000000.0f

typedef __attribute__((ext_vector_type(16))) _Float16 v16h;
typedef __attribute__((ext_vector_type(8)))  _Float16 v8h;
typedef __attribute__((ext_vector_type(8)))  float    v8f;

static __device__ __forceinline__ v16h combine8(v8h lo, v8h hi) {
    return __builtin_shufflevector(lo, hi, 0,1,2,3,4,5,6,7,8,9,10,11,12,13,14,15);
}

static __device__ __forceinline__ unsigned int pkh2(float a, float b) {
    union { _Float16 h[2]; unsigned int u; } x;
    x.h[0] = (_Float16)a; x.h[1] = (_Float16)b;
    return x.u;
}

// ---------------------------------------------------------------------------
// Prep: convert (K,CIN,COUT) f32 weights -> f16 in the exact WMMA B-matrix
// per-lane layout for stage 2. Element for (step s, n-tile t, lane L, vgpr v,
// half hh): row r = 32s + (L<16?0:16) + 2v + hh (r = k*64+c), col o = 16t+L%16.
// Rows r >= 960 (k == 15 padding) are zero.
// ---------------------------------------------------------------------------
__global__ void kpconv_prep_weights(const float* __restrict__ W,
                                    _Float16* __restrict__ wsh) {
    int gid = blockIdx.x * 256 + threadIdx.x;          // 32*4*32*8 = 32768
    if (gid >= 32 * 4 * 32 * 8) return;
    int v    = gid & 7;
    int lane = (gid >> 3) & 31;
    int t    = (gid >> 8) & 3;
    int s    = gid >> 10;
    int o    = t * 16 + (lane & 15);
    int rb   = s * 32 + ((lane < 16) ? 0 : 16) + 2 * v;
#pragma unroll
    for (int hh = 0; hh < 2; ++hh) {
        int r = rb + hh;
        float val = (r < KP * CIN) ? W[r * COUT + o] : 0.0f;
        wsh[gid * 2 + hh] = (_Float16)val;
    }
}

// ---------------------------------------------------------------------------
// Main fused kernel: one workgroup (128 threads / 4 waves) per 16 queries.
// ---------------------------------------------------------------------------
__global__ void __launch_bounds__(128)
kpconv_kernel(const float* __restrict__ s_feats,
              const float* __restrict__ q_points,
              const float* __restrict__ s_points,
              const int*   __restrict__ nbr,
              const float* __restrict__ kpnts,
              const _Float16* __restrict__ wsh,   // prepped weights (stage-2 B)
              float* __restrict__ out,
              int M, int N) {
    extern __shared__ __align__(32) char smem[];
    // [q][c][j] packed f16 pairs over h (B layout, stage 1): 16*64*16 u32 = 64KB
    unsigned int* featsT = (unsigned int*)smem;
    // [q][k][j] packed f16 pairs over h (A layout, stage 1): 16*16*16 u32 = 16KB
    unsigned int* nwT    = (unsigned int*)(smem + 65536);
    // [q][k*64+c] f16 weighted feats (stage-2 A): 16*1024 = 32KB
    _Float16*     wf     = (_Float16*)(smem + 81920);
    float*        kp     = (float*)(smem + 114688);   // 45 floats (pad 192B)
    float*        qp     = (float*)(smem + 114880);   // 48 floats
    int*          cnt    = (int*)(smem + 115072);     // 16 ints

    const int tid   = threadIdx.x;
    const int lane  = tid & 31;
    const int wv    = tid >> 5;
    const int qbase = blockIdx.x * QT;

    if (tid < QT) cnt[tid] = 0;
    if (tid < KP * 3) kp[tid] = kpnts[tid];
    if (tid < QT * 3) {
        int gq = qbase + tid / 3;
        qp[tid] = (gq < M) ? q_points[gq * 3 + (tid % 3)] : 0.0f;
    }
    __syncthreads();

    // ---------------- Phase 0: gather + convert + kernel weights -----------
    for (int task = tid; task < QT * (HN / 2); task += 128) {
        int q  = task >> 4;           // local query
        int j  = task & 15;           // neighbor pair index (h = 2j, 2j+1)
        int gq = qbase + q;
        bool qok = gq < M;
        int i0 = 0, i1 = 0;
        if (qok) {
            i0 = nbr[gq * HN + 2 * j];
            i1 = nbr[gq * HN + 2 * j + 1];
        }
        bool v0 = qok && ((unsigned)i0 < (unsigned)N);
        bool v1 = qok && ((unsigned)i1 < (unsigned)N);
        const float4* fp0 = (const float4*)(s_feats + (size_t)(v0 ? i0 : 0) * CIN);
        const float4* fp1 = (const float4*)(s_feats + (size_t)(v1 ? i1 : 0) * CIN);
        float m0 = v0 ? 1.0f : 0.0f, m1 = v1 ? 1.0f : 0.0f;
        float sum0 = 0.0f, sum1 = 0.0f;
        unsigned int* dst = featsT + q * 1024 + j;
#pragma unroll 4
        for (int cq = 0; cq < CIN / 4; ++cq) {
            float4 a = fp0[cq]; float4 b = fp1[cq];
            a.x *= m0; a.y *= m0; a.z *= m0; a.w *= m0;
            b.x *= m1; b.y *= m1; b.z *= m1; b.w *= m1;
            sum0 += a.x + a.y + a.z + a.w;
            sum1 += b.x + b.y + b.z + b.w;
            dst[(cq * 4 + 0) * 16] = pkh2(a.x, b.x);
            dst[(cq * 4 + 1) * 16] = pkh2(a.y, b.y);
            dst[(cq * 4 + 2) * 16] = pkh2(a.z, b.z);
            dst[(cq * 4 + 3) * 16] = pkh2(a.w, b.w);
        }
        float qx = qp[q * 3 + 0], qy = qp[q * 3 + 1], qz = qp[q * 3 + 2];
        float p0x = PT_INF, p0y = PT_INF, p0z = PT_INF;
        float p1x = PT_INF, p1y = PT_INF, p1z = PT_INF;
        if (v0) {
            p0x = s_points[(size_t)i0 * 3 + 0] - qx;
            p0y = s_points[(size_t)i0 * 3 + 1] - qy;
            p0z = s_points[(size_t)i0 * 3 + 2] - qz;
        }
        if (v1) {
            p1x = s_points[(size_t)i1 * 3 + 0] - qx;
            p1y = s_points[(size_t)i1 * 3 + 1] - qy;
            p1z = s_points[(size_t)i1 * 3 + 2] - qz;
        }
        unsigned int* nd = nwT + q * 256 + j;
#pragma unroll
        for (int k = 0; k < KP; ++k) {
            float kx = kp[k * 3 + 0], ky = kp[k * 3 + 1], kz = kp[k * 3 + 2];
            float dx0 = p0x - kx, dy0 = p0y - ky, dz0 = p0z - kz;
            float dx1 = p1x - kx, dy1 = p1y - ky, dz1 = p1z - kz;
            float sq0 = dx0 * dx0 + dy0 * dy0 + dz0 * dz0;
            float sq1 = dx1 * dx1 + dy1 * dy1 + dz1 * dz1;
            float w0 = fmaxf(0.0f, 1.0f - __builtin_sqrtf(sq0) * INV_SIGMA);
            float w1 = fmaxf(0.0f, 1.0f - __builtin_sqrtf(sq1) * INV_SIGMA);
            nd[k * 16] = pkh2(w0, w1);
        }
        nd[15 * 16] = 0u;   // zero-pad kernel row 15
        int add = (int)(v0 && (sum0 > 0.0f)) + (int)(v1 && (sum1 > 0.0f));
        atomicAdd(&cnt[q], add);
    }
    __syncthreads();

    // ---------------- Phase 1: per-query (16x32)@(32x64) via WMMA ----------
    // wave wv handles queries 4*wv .. 4*wv+3; one wmma per 16-col CIN tile.
    {
        const int arow = lane & 15;                 // A row m = kernel point k
        const int j0   = (lane >> 4) * 4;           // h-pair offset (0 or 4)
        const int hj0  = (lane < 16) ? 0 : 4;       // B h-pair offset
        const int kr0  = (lane >> 4) * 8;           // D row base
#pragma unroll
        for (int ql = 0; ql < 4; ++ql) {
            int q = wv * 4 + ql;
            const unsigned int* ab = nwT + q * 256 + arow * 16;
            v8h alo = *(const v8h*)(ab + j0);
            v8h ahi = *(const v8h*)(ab + j0 + 8);
            v16h A = combine8(alo, ahi);
            _Float16* wq = wf + q * 1024;
#pragma unroll
            for (int t = 0; t < 4; ++t) {
                const unsigned int* bb =
                    featsT + q * 1024 + (t * 16 + (lane & 15)) * 16;
                v8h blo = *(const v8h*)(bb + hj0);
                v8h bhi = *(const v8h*)(bb + hj0 + 8);
                v16h B = combine8(blo, bhi);
                v8f c = {0.f, 0.f, 0.f, 0.f, 0.f, 0.f, 0.f, 0.f};
                v8f d = __builtin_amdgcn_wmma_f32_16x16x32_f16(
                    false, A, false, B, (short)0, c, false, false);
                int ccol = t * 16 + (lane & 15);
#pragma unroll
                for (int v = 0; v < 8; ++v)
                    wq[(kr0 + v) * 64 + ccol] = (_Float16)d[v];
            }
        }
    }
    __syncthreads();

    // ---------------- Phase 2: (16x1024)@(1024x64) via WMMA chain ----------
    // wave wv owns output-column tile t = wv; 32 K-steps of 32.
    {
        const int t  = wv;
        const int rb = (lane >> 4) * 16;            // A col-sub offset
        const _Float16* aq = wf + (lane & 15) * 1024;
        v8f acc = {0.f, 0.f, 0.f, 0.f, 0.f, 0.f, 0.f, 0.f};
#pragma unroll 4
        for (int s = 0; s < 32; ++s) {
            const _Float16* ap = aq + s * 32 + rb;
            v8h alo = *(const v8h*)(ap);
            v8h ahi = *(const v8h*)(ap + 8);
            v16h A = combine8(alo, ahi);
            const _Float16* bp = wsh + (((size_t)(s * 4 + t) * 32 + lane) << 4);
            v8h blo = *(const v8h*)(bp);
            v8h bhi = *(const v8h*)(bp + 8);
            v16h B = combine8(blo, bhi);
            acc = __builtin_amdgcn_wmma_f32_16x16x32_f16(
                false, A, false, B, (short)0, acc, false, false);
        }
        int o  = t * 16 + (lane & 15);
        int q0 = (lane >> 4) * 8;
#pragma unroll
        for (int v = 0; v < 8; ++v) {
            int gq = qbase + q0 + v;
            if (gq < M) {
                int c_ = cnt[q0 + v];
                float inv = 1.0f / (float)(c_ < 1 ? 1 : c_);
                out[(size_t)gq * COUT + o] = acc[v] * inv;
            }
        }
    }
}

extern "C" void kernel_launch(void* const* d_in, const int* in_sizes, int n_in,
                              void* d_out, int out_size, void* d_ws, size_t ws_size,
                              hipStream_t stream) {
    const float* s_feats  = (const float*)d_in[0];
    const float* q_points = (const float*)d_in[1];
    const float* s_points = (const float*)d_in[2];
    const int*   nbr      = (const int*)d_in[3];
    const float* weights  = (const float*)d_in[4];
    const float* kpnts    = (const float*)d_in[5];
    float*       out      = (float*)d_out;

    const int N = in_sizes[0] / CIN;
    const int M = in_sizes[1] / 3;

    _Float16* wsh = (_Float16*)d_ws;   // needs 131072 bytes

    // Stage 0: pre-swizzle weight tensor into WMMA B layout (f16).
    kpconv_prep_weights<<<128, 256, 0, stream>>>(weights, wsh);

    // Fused gather + dual-GEMM kernel: 16 queries per workgroup.
    const int grid = (M + QT - 1) / QT;
    const size_t smem_bytes = 115136;   // 64K featsT + 16K nwT + 32K wf + misc
    kpconv_kernel<<<grid, 128, smem_bytes, stream>>>(
        s_feats, q_points, s_points, nbr, kpnts, wsh, out, M, N);
}
